// BiLSTM_CRF_44006234914987
// MI455X (gfx1250) — matrix operand order repaired and checked by reading, not compiled, not run on hardware
//
#include <hip/hip_runtime.h>
#include <hip/hip_bf16.h>
#include <math.h>

typedef __attribute__((ext_vector_type(16))) __bf16 v16bf;
typedef __attribute__((ext_vector_type(8)))  __bf16 v8bf;
typedef __attribute__((ext_vector_type(8)))  float  v8f;

#define SEQ   4096
#define EMBD  300
#define HD    150      // per-direction hidden
#define GATES 600      // 4*HD
#define NCOL  1200     // both directions' gates
#define NPAD  1216     // NCOL padded to multiple of 64
#define KPAD  320      // 300 padded to multiple of 32
#define NTAGS 6
#define STARTT 4
#define STOPT  5
#define NEGV  (-10000.0f)

// ---------------------------------------------------------------------------
// Prep 1: gather + convert X to zero-padded bf16  Xbf[SEQ][KPAD]
// ---------------------------------------------------------------------------
__global__ __launch_bounds__(KPAD)
void prep_x(const int* __restrict__ sent, const float* __restrict__ emb,
            __bf16* __restrict__ Xbf)
{
    const int t = blockIdx.x;
    const int k = threadIdx.x;
    float v = (k < EMBD) ? emb[(size_t)sent[t] * EMBD + k] : 0.0f;
    Xbf[(size_t)t * KPAD + k] = (__bf16)v;
}

// ---------------------------------------------------------------------------
// Prep 2: convert/pad weights to bf16 Wbf[NPAD][KPAD]; fuse biases biasP[NPAD]
// rows 0..599 = Wih_f, 600..1199 = Wih_b, 1200..1215 = zero
// ---------------------------------------------------------------------------
__global__ __launch_bounds__(KPAD)
void prep_w(const float* __restrict__ Wf, const float* __restrict__ Wb,
            const float* __restrict__ bihf, const float* __restrict__ bhhf,
            const float* __restrict__ bihb, const float* __restrict__ bhhb,
            __bf16* __restrict__ Wbf, float* __restrict__ biasP)
{
    const int n = blockIdx.x;
    const int k = threadIdx.x;
    float v = 0.0f;
    if (k < EMBD) {
        if (n < GATES)     v = Wf[(size_t)n * EMBD + k];
        else if (n < NCOL) v = Wb[(size_t)(n - GATES) * EMBD + k];
    }
    Wbf[(size_t)n * KPAD + k] = (__bf16)v;
    if (k == 0) {
        float b = 0.0f;
        if (n < GATES)     b = bihf[n] + bhhf[n];
        else if (n < NCOL) b = bihb[n - GATES] + bhhb[n - GATES];
        biasP[n] = b;
    }
}

// ---------------------------------------------------------------------------
// Phase 1: P = Xbf @ Wbf^T + bias via WMMA bf16 (f32 accumulate).
// One wave per block; wave computes a 16(M) x 64(N) tile. All loads are
// unguarded, 16B-aligned global_load_b128; only the final store has a
// wave-uniform column guard.
// ---------------------------------------------------------------------------
__global__ __launch_bounds__(32)
void proj_wmma(const __bf16* __restrict__ Xbf,
               const __bf16* __restrict__ Wbf,
               const float*  __restrict__ biasP,
               float* __restrict__ P)             // [SEQ, NCOL]
{
    const int lane  = threadIdx.x;
    const int half  = lane >> 4;     // 0 or 1
    const int l16   = lane & 15;
    const int mBase = blockIdx.x * 16;
    const int nBase = blockIdx.y * 64;

    // A fragment: lane (half,l16) holds row M=l16,
    // K = kb + half*8 + {0..7} and kb + half*8 + 16 + {0..7}
    const __bf16* __restrict__ aPtr = Xbf + (size_t)(mBase + l16) * KPAD + half * 8;

    v8f acc[4] = {};

    for (int kb = 0; kb < KPAD; kb += 32) {
        v8bf a0 = *(const v8bf*)(aPtr + kb);
        v8bf a1 = *(const v8bf*)(aPtr + kb + 16);
        v16bf a;
#pragma unroll
        for (int i = 0; i < 8; ++i) { a[i] = a0[i]; a[8 + i] = a1[i]; }

#pragma unroll
        for (int s = 0; s < 4; ++s) {
            // B fragment: lane holds column N = nBase+s*16+l16,
            // K = kb + half*8 + {0..7} and kb + half*8 + 16 + {0..7}
            const __bf16* __restrict__ bPtr =
                Wbf + (size_t)(nBase + s * 16 + l16) * KPAD + half * 8 + kb;
            v8bf b0 = *(const v8bf*)(bPtr);
            v8bf b1 = *(const v8bf*)(bPtr + 16);
            v16bf b;
#pragma unroll
            for (int i = 0; i < 8; ++i) { b[i] = b0[i]; b[8 + i] = b1[i]; }

            acc[s] = __builtin_amdgcn_wmma_f32_16x16x32_bf16(
                false, a, false, b, (short)0, acc[s], false, false);
        }
    }

    // C/D layout: VGPR v -> M = v + half*8 ; N = l16. Uniform column guard.
#pragma unroll
    for (int s = 0; s < 4; ++s) {
        if (nBase + s * 16 < NCOL) {                 // wave-uniform
            const int col = nBase + s * 16 + l16;
            const float bias = biasP[col];
#pragma unroll
            for (int v = 0; v < 8; ++v) {
                int m = mBase + v + half * 8;
                P[(size_t)m * NCOL + col] = acc[s][v] + bias;
            }
        }
    }
}

// ---------------------------------------------------------------------------
// Phase 2: sequential LSTM recurrence, one block per direction.
// 608 threads: threads 0..599 compute gate preactivations (+ activation),
// threads 0..149 update (c, h). h broadcast via LDS; Whh streamed from L2.
// ---------------------------------------------------------------------------
__global__ __launch_bounds__(608)
void bilstm_recurrence(const float* __restrict__ P,
                       const float* __restrict__ Whh_f,
                       const float* __restrict__ Whh_b,
                       float* __restrict__ hs)           // [SEQ, 300]
{
    const int dir = blockIdx.x;          // 0 = forward, 1 = backward
    const int tid = threadIdx.x;
    const float* __restrict__ Whh = dir ? Whh_b : Whh_f;

    __shared__ float h_sh[160];
    __shared__ float g_sh[608];

    if (tid < 160) h_sh[tid] = 0.0f;
    float c = 0.0f;
    __syncthreads();

    for (int s = 0; s < SEQ; ++s) {
        const int t = dir ? (SEQ - 1 - s) : s;

        if (tid < GATES) {
            float acc = P[(size_t)t * NCOL + dir * GATES + tid];
            const float* __restrict__ w = Whh + (size_t)tid * HD;
#pragma unroll 10
            for (int k = 0; k < HD; ++k) acc += w[k] * h_sh[k];
            const int gate = tid / HD;   // 0=i 1=f 2=g 3=o
            g_sh[tid] = (gate == 2) ? tanhf(acc)
                                    : (1.0f / (1.0f + __expf(-acc)));
        }
        __syncthreads();

        if (tid < HD) {
            c = g_sh[HD + tid] * c + g_sh[tid] * g_sh[2 * HD + tid];
            float h = g_sh[3 * HD + tid] * tanhf(c);
            h_sh[tid] = h;
            hs[(size_t)t * (2 * HD) + dir * HD + tid] = h;
        }
        __syncthreads();
    }
}

// ---------------------------------------------------------------------------
// Phase 3: feats[t] = W_out @ lstm_out[t] + b_out   (one thread per timestep)
// ---------------------------------------------------------------------------
__global__ __launch_bounds__(256)
void bilstm_feats(const float* __restrict__ hs,
                  const float* __restrict__ Wout,        // [6,300]
                  const float* __restrict__ bout,        // [6]
                  float* __restrict__ feats)             // [SEQ,6]
{
    const int t = blockIdx.x * blockDim.x + threadIdx.x;
    if (t >= SEQ) return;
    const float* __restrict__ h = hs + (size_t)t * (2 * HD);
#pragma unroll
    for (int j = 0; j < NTAGS; ++j) {
        float acc = bout[j];
        const float* __restrict__ w = Wout + j * (2 * HD);
#pragma unroll 10
        for (int k = 0; k < 2 * HD; ++k) acc += w[k] * h[k];
        feats[t * NTAGS + j] = acc;
    }
}

// ---------------------------------------------------------------------------
// Phase 4: Viterbi scan (single wave) + backtrace (single lane).
// fv held one tag per lane (lanes 0..5); shfl broadcast of prev-tag scores.
// Backpointers in LDS (4096*6 bytes). Output: out[0]=score, out[1..4096]=path.
// ---------------------------------------------------------------------------
__global__ __launch_bounds__(32)
void crf_viterbi(const float* __restrict__ feats,
                 const float* __restrict__ trans,        // [6,6] trans[next][prev]
                 float* __restrict__ out)
{
    __shared__ unsigned char bp[SEQ][NTAGS];
    __shared__ float term_sh[NTAGS];

    const int lane = threadIdx.x;
    float fv = (lane == STARTT) ? 0.0f : NEGV;

    float tr[NTAGS];
    if (lane < NTAGS) {
#pragma unroll
        for (int p = 0; p < NTAGS; ++p) tr[p] = trans[lane * NTAGS + p];
    }

    for (int t = 0; t < SEQ; ++t) {
        float best = -3.4e38f;
        int   barg = 0;
#pragma unroll
        for (int p = 0; p < NTAGS; ++p) {
            float fvp = __shfl(fv, p, 32);               // uniform: all lanes execute
            if (lane < NTAGS) {
                float sc = fvp + tr[p];
                if (sc > best) { best = sc; barg = p; }  // '>' keeps first max (argmax)
            }
        }
        if (lane < NTAGS) {
            fv = best + feats[t * NTAGS + lane];
            bp[t][lane] = (unsigned char)barg;
        }
    }

    if (lane < NTAGS) term_sh[lane] = fv + trans[STOPT * NTAGS + lane];
    __syncthreads();

    if (lane == 0) {
        int   last = 0;
        float best = term_sh[0];
#pragma unroll
        for (int j = 1; j < NTAGS; ++j)
            if (term_sh[j] > best) { best = term_sh[j]; last = j; }
        out[0] = best;
        int tag = last;
        for (int t = SEQ - 1; t >= 1; --t) {
            out[1 + t] = (float)tag;
            tag = bp[t][tag];
        }
        out[1] = (float)tag;
    }
}

// ---------------------------------------------------------------------------
extern "C" void kernel_launch(void* const* d_in, const int* in_sizes, int n_in,
                              void* d_out, int out_size, void* d_ws, size_t ws_size,
                              hipStream_t stream) {
    const int*   sentence = (const int*)  d_in[0];
    const float* emb      = (const float*)d_in[1];
    const float* Wih_f    = (const float*)d_in[2];
    const float* Whh_f    = (const float*)d_in[3];
    const float* bih_f    = (const float*)d_in[4];
    const float* bhh_f    = (const float*)d_in[5];
    const float* Wih_b    = (const float*)d_in[6];
    const float* Whh_b    = (const float*)d_in[7];
    const float* bih_b    = (const float*)d_in[8];
    const float* bhh_b    = (const float*)d_in[9];
    const float* W_out    = (const float*)d_in[10];
    const float* b_out    = (const float*)d_in[11];
    const float* trans    = (const float*)d_in[12];

    // Workspace layout (all 16B aligned)
    float*  P     = (float*)d_ws;                        // [SEQ, NCOL]
    float*  hs    = P  + (size_t)SEQ * NCOL;             // [SEQ, 300]
    float*  feats = hs + (size_t)SEQ * (2 * HD);         // [SEQ, 6]
    __bf16* Xbf   = (__bf16*)(feats + (size_t)SEQ * NTAGS);   // [SEQ, KPAD]
    __bf16* Wbf   = Xbf + (size_t)SEQ * KPAD;                 // [NPAD, KPAD]
    float*  biasP = (float*)(Wbf + (size_t)NPAD * KPAD);      // [NPAD]

    prep_x<<<SEQ, KPAD, 0, stream>>>(sentence, emb, Xbf);
    prep_w<<<NPAD, KPAD, 0, stream>>>(Wih_f, Wih_b, bih_f, bhh_f, bih_b, bhh_b,
                                      Wbf, biasP);

    dim3 gGemm(SEQ / 16, NPAD / 64);                     // 256 x 19
    proj_wmma<<<gGemm, 32, 0, stream>>>(Xbf, Wbf, biasP, P);

    bilstm_recurrence<<<2, 608, 0, stream>>>(P, Whh_f, Whh_b, hs);

    bilstm_feats<<<(SEQ + 255) / 256, 256, 0, stream>>>(hs, W_out, b_out, feats);

    crf_viterbi<<<1, 32, 0, stream>>>(feats, trans, (float*)d_out);
}
